// WindowAttention_61658550501924
// MI455X (gfx1250) — matrix operand order repaired and checked by reading, not compiled, not run on hardware
//
#include <hip/hip_runtime.h>
#include <hip/hip_bf16.h>
#include <math.h>

// ---------------- problem constants ----------------
#define DIMC   192
#define NH     6
#define HDIM   32
#define NTOK   64
#define NWIN   64
#define NBATCH 2048
#define QSCALE 0.17677669529663687f   // 32^-0.5

typedef __attribute__((ext_vector_type(16))) _Float16 v16h;
typedef __attribute__((ext_vector_type(8)))  _Float16 v8h;
typedef __attribute__((ext_vector_type(4)))  _Float16 v4h;
typedef __attribute__((ext_vector_type(8)))  float    v8f;

// ---------------- workspace (d_ws) byte offsets ----------------
// qwP  f16 packed frags [6kt][12nt][32 lane][16]   36,864 h =  73,728 B
// kvwP f16 packed frags [6kt][24nt][32][16]        73,728 h = 147,456 B
// pwP  f16 packed frags [6kt][12nt][32][16]        36,864 h =  73,728 B
// biasTab f32 [6][64][64]                                      98,304 B
#define WS_QWP   0
#define WS_KVWP  73728
#define WS_PWP   221184
#define WS_BIAS  294912

// ---------------- LDS carve (bytes) ----------------
// @0      xh1 f16[64][200] 25600   | phase2: yf f16[32][392] 25088 ;
//                                  |         en f32[32][33] @25088 ; attb f16[32][40] @29312
// @25600  xh2 f16[64][200] 25600
// @51200  qh  f16[64][200] 25600
// @76800  kh  f16[6][64][40] 30720 | later: sbuf f16[64][200] 25600
// @107520 vhT f16[6][32][72] 27648
// @135168 yfT f16[384][40]   30720
// @165888 sc  f32[64][65]    16640
// @182528 pr  f16[64][72]     9216   -> total 191,744
#define SMEM_BYTES 191744

__device__ __forceinline__ v16h join8(v8h lo, v8h hi) {
  return __builtin_shufflevector(lo, hi, 0, 1, 2, 3, 4, 5, 6, 7, 8, 9, 10, 11,
                                 12, 13, 14, 15);
}

// ---------------- CDNA5 wave32 WMMA fragment loaders (all 16B vector ops) ----
// A 16x32 f16: lane m = lane&15; lanes<16 K=0..15, lanes>=16 K=16..31
// -> 16 contiguous halfs (two 8-half chunks at p and p+8). ldm must be a multiple of 8.
__device__ __forceinline__ v16h frag_a(const _Float16* base, int row0, int ldm,
                                       int k0, int lane) {
  const _Float16* p =
      base + (row0 + (lane & 15)) * ldm + k0 + ((lane & 16) ? 16 : 0);
  return join8(*(const v8h*)p, *(const v8h*)(p + 8));
}

// B 32x16 f16 where memory holds M[n][k] (row stride ldm): B[k][n] = M[n][k].
// lane n = lane&15; lanes<16 K=0..15, lanes>=16 K=16..31 -> 16 contiguous halfs.
__device__ __forceinline__ v16h frag_b_col(const _Float16* base, int k0, int ldm,
                                           int col0, int lane) {
  const _Float16* p =
      base + (col0 + (lane & 15)) * ldm + k0 + ((lane & 16) ? 16 : 0);
  return join8(*(const v8h*)p, *(const v8h*)(p + 8));
}

// B fragment from pre-packed global weights: one aligned 32-byte load per lane.
__device__ __forceinline__ v16h frag_b_g(const _Float16* packed, int ntiles,
                                         int kt, int nt, int lane) {
  return *(const v16h*)(packed + (((kt * ntiles + nt) * 32 + lane) << 4));
}

__device__ __forceinline__ v8f wmma16(v16h a, v16h b, v8f c) {
  return __builtin_amdgcn_wmma_f32_16x16x32_f16(false, a, false, b, (short)0, c,
                                                false, false);
}

// ---------------- prep: pack weights into B-fragment order + bias gather ----
__global__ __launch_bounds__(256) void wattn_prep(
    const float* __restrict__ qw, const float* __restrict__ kvw,
    const float* __restrict__ pw, const float* __restrict__ rpb,
    const int* __restrict__ rel_idx, _Float16* __restrict__ qwP,
    _Float16* __restrict__ kvwP, _Float16* __restrict__ pwP,
    float* __restrict__ biasTab) {
  int idx = blockIdx.x * 256 + threadIdx.x;
  if (idx < 36864) {                       // qwP: NT=12 tiles of 16 cols
    int i = idx & 15, l = (idx >> 4) & 31, tile = idx >> 9;
    int nt = tile % 12, kt = tile / 12;
    int k = kt * 32 + ((l >> 4) & 1) * 16 + i;
    int n = nt * 16 + (l & 15);
    qwP[idx] = (_Float16)qw[n * DIMC + k];
  } else if (idx < 110592) {               // kvwP: NT=24
    int local = idx - 36864;
    int i = local & 15, l = (local >> 4) & 31, tile = local >> 9;
    int nt = tile % 24, kt = tile / 24;
    int k = kt * 32 + ((l >> 4) & 1) * 16 + i;
    int n = nt * 16 + (l & 15);
    kvwP[local] = (_Float16)kvw[n * DIMC + k];
  } else if (idx < 147456) {               // pwP: NT=12
    int local = idx - 110592;
    int i = local & 15, l = (local >> 4) & 31, tile = local >> 9;
    int nt = tile % 12, kt = tile / 12;
    int k = kt * 32 + ((l >> 4) & 1) * 16 + i;
    int n = nt * 16 + (l & 15);
    pwP[local] = (_Float16)pw[n * DIMC + k];
  } else if (idx < 172032) {               // biasTab[h][n*64+m] = rpb[rel_idx][h]
    int l = idx - 147456;
    int h = l / 4096, r = l % 4096;
    biasTab[h * 4096 + r] = rpb[rel_idx[r] * NH + h];
  }
}

// ---------------- fused window-attention + LAM + projection ----------------
__global__ __launch_bounds__(256, 1) void wattn_main(
    const float* __restrict__ x1, const float* __restrict__ x2,
    const float* __restrict__ mask, const float* __restrict__ qb,
    const float* __restrict__ kvb, const float* __restrict__ gamma_p,
    const float* __restrict__ pb, const _Float16* __restrict__ qwP,
    const _Float16* __restrict__ kvwP, const _Float16* __restrict__ pwP,
    const float* __restrict__ biasTab, float* __restrict__ out) {
  extern __shared__ char smem[];
  _Float16* xh1  = (_Float16*)(smem + 0);       // [64][200]
  _Float16* xh2  = (_Float16*)(smem + 25600);   // [64][200]
  _Float16* yf   = (_Float16*)(smem + 0);       // [32][392] (reuses xh1/xh2)
  float*    en   = (float*)   (smem + 25088);   // [32][33]
  _Float16* attb = (_Float16*)(smem + 29312);   // [32][40]
  _Float16* qh   = (_Float16*)(smem + 51200);   // [64][200]
  _Float16* kh   = (_Float16*)(smem + 76800);   // [6][64][40]
  _Float16* sbuf = (_Float16*)(smem + 76800);   // [64][200] (reuses kh)
  _Float16* vhT  = (_Float16*)(smem + 107520);  // [6][32][72]  v transposed
  _Float16* yfT  = (_Float16*)(smem + 135168);  // [384][40]    flat transposed
  float*    sc   = (float*)   (smem + 165888);  // [64][65]
  _Float16* pr   = (_Float16*)(smem + 182528);  // [64][72]

  const int b    = blockIdx.x;
  const int tid  = threadIdx.x;
  // wave id as an SGPR: all tile indices / store-target selections derived from
  // it become scalar (no exec-mask branching around WMMA regions)
  const int wid  = __builtin_amdgcn_readfirstlane(tid >> 5);
  const int lane = tid & 31;
  const int w    = b & (NWIN - 1);
  const float gamma = gamma_p[0];

  const float* x1b = x1 + (size_t)b * NTOK * DIMC;
  const float* x2b = x2 + (size_t)b * NTOK * DIMC;

  __builtin_prefetch(qwP, 0, 1);
  __builtin_prefetch(kvwP, 0, 1);

  // ---- stage x1/x2 -> LDS f16 (packed 8B stores) ----
  for (int e = tid; e < NTOK * DIMC / 4; e += 256) {
    const float4 a1 = ((const float4*)x1b)[e];
    const float4 a2 = ((const float4*)x2b)[e];
    int n = (e * 4) / DIMC, c = (e * 4) % DIMC;
    v4h h1 = {(_Float16)a1.x, (_Float16)a1.y, (_Float16)a1.z, (_Float16)a1.w};
    v4h h2 = {(_Float16)a2.x, (_Float16)a2.y, (_Float16)a2.z, (_Float16)a2.w};
    *(v4h*)(xh1 + n * 200 + c) = h1;
    *(v4h*)(xh2 + n * 200 + c) = h2;
  }
  __syncthreads();

  const int mlb = (lane & 16) ? 8 : 0;   // C/D row offset for this half-wave
  const int nl  = lane & 15;             // C/D column within tile

  // ---- q = (x1 @ qw^T + qb) * SCALE  -> qh f16 (pre-scaled) ----
  for (int i = 0; i < 6; ++i) {
    int t = wid + 8 * i, tm = t / 12, tn = t % 12;
    v8f acc = {};
#pragma unroll
    for (int kk = 0; kk < 6; ++kk)
      acc = wmma16(frag_a(xh1, tm * 16, 200, kk * 32, lane),
                   frag_b_g(qwP, 12, kk, tn, lane), acc);
    int nc = tn * 16 + nl;
    float bias = qb[nc];
#pragma unroll
    for (int r = 0; r < 8; ++r)
      qh[(tm * 16 + r + mlb) * 200 + nc] = (_Float16)((acc[r] + bias) * QSCALE);
  }

  // ---- kv = x2 @ kvw^T + kvb -> kh [h][n][d] / vhT [h][d][n] ----
  // Tile (tm,tn): tn<12 -> K half, tn>=12 -> V half. head = (tn%12)>>1,
  // d-offset = ((tn%12)&1)*16 -- all wave-uniform (SGPR) selections.
  for (int i = 0; i < 12; ++i) {
    int t = wid + 8 * i, tm = t / 24, tn = t % 24;
    v8f acc = {};
#pragma unroll
    for (int kk = 0; kk < 6; ++kk)
      acc = wmma16(frag_a(xh2, tm * 16, 200, kk * 32, lane),
                   frag_b_g(kvwP, 24, kk, tn, lane), acc);
    int tloc = (tn >= 12) ? (tn - 12) : tn;   // scalar
    int h    = tloc >> 1;                     // scalar head index
    int d    = ((tloc & 1) << 4) + nl;        // per-lane head-dim index
    float bias = kvb[tn * 16 + nl];
    if (tn < 12) {                            // scalar branch: K half
      _Float16* dst = kh + h * (NTOK * 40) + d;
#pragma unroll
      for (int r = 0; r < 8; ++r)
        dst[(tm * 16 + r + mlb) * 40] = (_Float16)(acc[r] + bias);
    } else {                                  // scalar branch: V half
      _Float16* dst = vhT + h * (HDIM * 72) + d * 72;
#pragma unroll
      for (int r = 0; r < 8; ++r)
        dst[tm * 16 + r + mlb] = (_Float16)(acc[r] + bias);
    }
  }
  __syncthreads();

  // ---- per-head attention ----
  for (int h = 0; h < NH; ++h) {
    // scores = q_h @ k_h^T + bias + mask   (16 tiles, K=32)
#pragma unroll
    for (int i = 0; i < 2; ++i) {
      int t = wid + 8 * i, tm = t >> 2, tn = t & 3;
      v8f acc = {};
      acc = wmma16(frag_a(qh, tm * 16, 200, h * 32, lane),
                   frag_b_col(kh + h * NTOK * 40, 0, 40, tn * 16, lane), acc);
      int mc = tn * 16 + nl;
#pragma unroll
      for (int r = 0; r < 8; ++r) {
        int mr = tm * 16 + r + mlb;
        sc[mr * 65 + mc] = acc[r] + biasTab[h * 4096 + mr * 64 + mc] +
                           mask[w * 4096 + mr * 64 + mc];
      }
    }
    __syncthreads();
    // softmax over keys (rows of 64)
    if (tid < 64) {
      float mx = -1e30f;
      for (int j = 0; j < 64; ++j) mx = fmaxf(mx, sc[tid * 65 + j]);
      float s = 0.f;
      for (int j = 0; j < 64; ++j) s += __expf(sc[tid * 65 + j] - mx);
      float inv = 1.f / s;
      for (int j = 0; j < 64; ++j)
        pr[tid * 72 + j] = (_Float16)(__expf(sc[tid * 65 + j] - mx) * inv);
    }
    __syncthreads();
    // y_h = probs @ v_h  (8 tiles, K=64); write both flat layouts
    {
      int tm = wid >> 1, tn = wid & 1;
      v8f acc = {};
#pragma unroll
      for (int kk = 0; kk < 2; ++kk)
        acc = wmma16(
            frag_a(pr, tm * 16, 72, kk * 32, lane),
            frag_b_col(vhT + h * HDIM * 72, kk * 32, 72, tn * 16, lane), acc);
      int d = tn * 16 + nl;
#pragma unroll
      for (int r = 0; r < 8; ++r) {
        int nr = tm * 16 + r + mlb;
        _Float16 v = (_Float16)acc[r];
        yf[d * 392 + h * 64 + nr] = v;          // flat[d][h*64+n]
        yfT[(h * 64 + nr) * 40 + d] = v;        // flat^T[c][d]
      }
    }
    __syncthreads();
  }

  // ---- LAM: energy = flat @ flat^T (waves 0-3) || residual init (waves 4-7) --
  if (wid < 4) {                                // scalar branch
    int tm = wid >> 1, tn = wid & 1;
    v8f acc = {};
#pragma unroll
    for (int kk = 0; kk < 12; ++kk)
      acc = wmma16(frag_a(yf, tm * 16, 392, kk * 32, lane),
                   frag_b_col(yf, kk * 32, 392, tn * 16, lane), acc);
    int jc = tn * 16 + nl;
#pragma unroll
    for (int r = 0; r < 8; ++r)
      en[(tm * 16 + r + mlb) * 33 + jc] = acc[r];
  } else {
    // sbuf[n][c] = q_scaled + x1   (overwrites kh; attention done)
    for (int e4 = (wid - 4) * 32 + lane; e4 < NTOK * DIMC / 4; e4 += 128) {
      float4 xv = ((const float4*)x1b)[e4];
      int n = (e4 * 4) / DIMC, c = (e4 * 4) % DIMC;
      v4h qv = *(const v4h*)(qh + n * 200 + c);
      v4h sv = {(_Float16)((float)qv[0] + xv.x), (_Float16)((float)qv[1] + xv.y),
                (_Float16)((float)qv[2] + xv.z), (_Float16)((float)qv[3] + xv.w)};
      *(v4h*)(sbuf + n * 200 + c) = sv;
    }
  }
  __syncthreads();
  // channel softmax (raw energy, per reference)
  if (tid < 32) {
    float mx = -1e30f;
    for (int j = 0; j < 32; ++j) mx = fmaxf(mx, en[tid * 33 + j]);
    float s = 0.f;
    for (int j = 0; j < 32; ++j) s += __expf(en[tid * 33 + j] - mx);
    float inv = 1.f / s;
    for (int j = 0; j < 32; ++j)
      attb[tid * 40 + j] = (_Float16)(__expf(en[tid * 33 + j] - mx) * inv);
  }
  __syncthreads();
  // out = att @ flat ; s += gamma*out + flat   (48 tiles, K=32)
  for (int i = 0; i < 6; ++i) {
    int t = wid + 8 * i, tm = t / 24, tn = t % 24;
    v8f acc = {};
    acc = wmma16(frag_a(attb, tm * 16, 40, 0, lane),
                 frag_b_col(yfT, 0, 40, tn * 16, lane), acc);
    int cc = tn * 16 + nl;
    int hh = tn >> 2;                  // scalar: cc>>6 == tn>>2 (tile-aligned)
    int nn = cc & 63;
#pragma unroll
    for (int r = 0; r < 8; ++r) {
      int dd = tm * 16 + r + mlb;
      float o = gamma * acc[r] + (float)yfT[cc * 40 + dd];
      int si = nn * 200 + hh * 32 + dd;  // unique per lane/tile -> safe RMW
      sbuf[si] = (_Float16)((float)sbuf[si] + o);
    }
  }
  __syncthreads();

  // ---- final projection: out = sbuf @ pw^T + pb ----
  float* outb = out + (size_t)b * NTOK * DIMC;
  for (int i = 0; i < 6; ++i) {
    int t = wid + 8 * i, tm = t / 12, tn = t % 12;
    v8f acc = {};
#pragma unroll
    for (int kk = 0; kk < 6; ++kk)
      acc = wmma16(frag_a(sbuf, tm * 16, 200, kk * 32, lane),
                   frag_b_g(pwP, 12, kk, tn, lane), acc);
    int nc = tn * 16 + nl;
    float bias = pb[nc];
#pragma unroll
    for (int r = 0; r < 8; ++r)
      outb[(tm * 16 + r + mlb) * DIMC + nc] = acc[r] + bias;
  }
}

// ---------------- launch ----------------
extern "C" void kernel_launch(void* const* d_in, const int* in_sizes, int n_in,
                              void* d_out, int out_size, void* d_ws,
                              size_t ws_size, hipStream_t stream) {
  (void)in_sizes; (void)n_in; (void)out_size; (void)ws_size;
  const float* x1      = (const float*)d_in[0];
  const float* x2      = (const float*)d_in[1];
  const float* mask    = (const float*)d_in[2];
  const float* qw      = (const float*)d_in[3];
  const float* qb      = (const float*)d_in[4];
  const float* kvw     = (const float*)d_in[5];
  const float* kvb     = (const float*)d_in[6];
  const float* rpb     = (const float*)d_in[7];
  const float* gamma   = (const float*)d_in[8];
  const float* pw      = (const float*)d_in[9];
  const float* pb      = (const float*)d_in[10];
  const int*   rel_idx = (const int*)d_in[11];

  char* ws = (char*)d_ws;
  _Float16* qwP     = (_Float16*)(ws + WS_QWP);
  _Float16* kvwP    = (_Float16*)(ws + WS_KVWP);
  _Float16* pwP     = (_Float16*)(ws + WS_PWP);
  float*    biasTab = (float*)(ws + WS_BIAS);

  wattn_prep<<<672, 256, 0, stream>>>(qw, kvw, pw, rpb, rel_idx, qwP, kvwP,
                                      pwP, biasTab);
  wattn_main<<<NBATCH, 256, SMEM_BYTES, stream>>>(
      x1, x2, mask, qb, kvb, gamma, pb, qwP, kvwP, pwP, biasTab,
      (float*)d_out);
}
